// MoEProcessor_37546604102321
// MI455X (gfx1250) — matrix operand order repaired and checked by reading, not compiled, not run on hardware
//
#include <hip/hip_runtime.h>
#include <hip/hip_bf16.h>
#include <cstddef>

// ---------------------------------------------------------------------------
// MI455X (gfx1250): wave32 + v_wmma_f32_16x16x32_bf16 GEMMs, TDM tile staging.
// ---------------------------------------------------------------------------

typedef __bf16 bf16;
typedef __attribute__((ext_vector_type(16))) __bf16 v16bf;
typedef __attribute__((ext_vector_type(8)))  __bf16 v8bf;
typedef __attribute__((ext_vector_type(8)))  float  v8f;
typedef __attribute__((ext_vector_type(4))) unsigned int u32x4;
typedef __attribute__((ext_vector_type(8))) int i32x8;
typedef __attribute__((ext_vector_type(4))) int i32x4;

#define WMMA_BF16(a, b, c) \
  __builtin_amdgcn_wmma_f32_16x16x32_bf16(false, (a), false, (b), (short)0, (c), false, false)

#if __has_builtin(__builtin_amdgcn_tensor_load_to_lds) && __has_builtin(__builtin_amdgcn_s_wait_tensorcnt)
#define HAVE_TDM 1
#else
#define HAVE_TDM 0
#endif

static constexpr int Bc = 8, Nc = 4096, Dc = 256;
static constexpr int NHc = 8, DHc = 32, Ec = 4, TDc = 768, HIDc = 1024;

// -------------------- workspace layout (float offsets, then bf16 region) ---
static constexpr size_t OFF_TAB  = 0;          // 12288 floats of DFT tables
static constexpr size_t OFF_CW   = 12288;      // 32 floats (cw[B][E])
static constexpr size_t F_XFTR   = 12320;      // B*D*32*16 = 1048576
static constexpr size_t F_XFTI   = F_XFTR + 1048576;
static constexpr size_t F_OFTR   = F_XFTI + 1048576;
static constexpr size_t F_OFTI   = F_OFTR + 1048576;
static constexpr size_t F_X2     = F_OFTI + 1048576;   // 32768*256 fp32
static constexpr size_t F_END    = F_X2 + 8388608;
// bf16 region element offsets
static constexpr size_t H_HID = 0;                       // 32768*1024
static constexpr size_t H_QKV = H_HID + 33554432ull;     // 32768*768
static constexpr size_t H_LN  = H_QKV + 25165824ull;     // 32768*256
static constexpr size_t H_AO  = H_LN  + 8388608ull;      // 32768*256
static constexpr size_t H_XA  = H_AO  + 8388608ull;      // 32768*256

__device__ __forceinline__ float gelu_f(float x) {
  return 0.5f * x * (1.0f + erff(x * 0.70710678118654752f));
}

__device__ __forceinline__ v16bf mk16(v8bf lo, v8bf hi) {
  return __builtin_shufflevector(lo, hi, 0, 1, 2, 3, 4, 5, 6, 7,
                                 8, 9, 10, 11, 12, 13, 14, 15);
}

#if HAVE_TDM
// Issue a TDM 2-D tile load: tile_dim0=32 elems (K), tile_dim1=64 rows, into
// packed LDS at ldsDst.  D# layout per CDNA5 ISA 8.3/8.4.  6-arg builtin
// (clang-23 flavor): (g0, g1, g2, g3, g4, cpol).
template <typename T>
__device__ __forceinline__ void tdm_load_tile(T* ldsDst, const T* gsrc,
                                              int rowStride, int rowsLeft,
                                              int colsLeft) {
  u32x4 g0;
  i32x8 g1;
  i32x4 gz4;
  i32x8 gz8;
  gz4[0] = 0; gz4[1] = 0; gz4[2] = 0; gz4[3] = 0;
  gz8[0] = 0; gz8[1] = 0; gz8[2] = 0; gz8[3] = 0;
  gz8[4] = 0; gz8[5] = 0; gz8[6] = 0; gz8[7] = 0;
  const unsigned lds = (unsigned)(unsigned long long)(void*)ldsDst;  // LDS aperture: offset in addr[31:0]
  const unsigned long long ga = (unsigned long long)gsrc;
  g0[0] = 1u;                                           // count=1, user descriptor
  g0[1] = lds;                                          // lds_addr
  g0[2] = (unsigned)ga;                                 // global_addr[31:0]
  g0[3] = (unsigned)((ga >> 32) & 0x01FFFFFFull) | (2u << 30);  // addr[56:32] | type=2
  const unsigned dsz = (sizeof(T) == 4) ? 2u : 1u;      // 4B / 2B elements
  const unsigned td0 = (unsigned)colsLeft;              // tensor_dim0 (to edge)
  const unsigned td1 = (unsigned)rowsLeft;              // tensor_dim1
  const unsigned st0 = (unsigned)rowStride;             // tensor_dim0_stride
  g1[0] = (int)(dsz << 16);                             // wg_mask=0, data_size
  g1[1] = (int)((td0 & 0xFFFFu) << 16);                 // tensor_dim0[15:0]
  g1[2] = (int)((td0 >> 16) | ((td1 & 0xFFFFu) << 16)); // dim0[31:16], dim1[15:0]
  g1[3] = (int)((td1 >> 16) | (32u << 16));             // dim1[31:16], tile_dim0=32
  g1[4] = (int)64u;                                     // tile_dim1=64, tile_dim2=0
  g1[5] = (int)st0;                                     // stride0[31:0]
  g1[6] = 0;                                            // stride0[47:32], stride1 lo
  g1[7] = 0;
  __builtin_amdgcn_tensor_load_to_lds(g0, g1, gz4, gz4, gz8, 0);
}
#endif

// ------------------------------ DFT tables ---------------------------------
__global__ void fill_tables_k(float* __restrict__ tab) {
  const int tid = threadIdx.x;
  const float TP = 6.28318530717958648f;
  for (int i = tid; i < 1024; i += 256) {
    int w = i >> 4, ky = i & 15;
    float a = TP * (float)(w * ky) / 64.f;
    tab[i] = __cosf(a); tab[1024 + i] = -__sinf(a);
  }
  for (int i = tid; i < 2048; i += 256) {
    int slot = i >> 6, hh = i & 63;
    int kx = (slot < 16) ? slot : slot + 32;
    float a = TP * (float)(hh * kx) / 64.f;
    tab[2048 + i] = __cosf(a); tab[4096 + i] = -__sinf(a);
  }
  for (int i = tid; i < 2048; i += 256) {
    int hh = i >> 5, slot = i & 31;
    int kx = (slot < 16) ? slot : slot + 32;
    float a = TP * (float)(hh * kx) / 64.f;
    tab[6144 + i] = __cosf(a); tab[8192 + i] = __sinf(a);
  }
  for (int i = tid; i < 1024; i += 256) {
    int ky = i >> 6, w = i & 63;
    float c = (ky == 0) ? 1.f : 2.f;
    float a = TP * (float)(ky * w) / 64.f;
    tab[10240 + i] = c * __cosf(a); tab[11264 + i] = c * __sinf(a);
  }
}

// ------------------------------ gating -------------------------------------
__global__ __launch_bounds__(256) void gating_k(
    const float* __restrict__ x, const float* __restrict__ text,
    const float* __restrict__ w1, const float* __restrict__ b1,
    const float* __restrict__ w2, const float* __restrict__ b2,
    float* __restrict__ cw, float* __restrict__ wout) {
  const int b = blockIdx.x, tid = threadIdx.x;
  __shared__ float feat[Dc + TDc];
  __shared__ float hid[Dc];
  __shared__ float logits[Ec];
  float s = 0.f;
  const float* xb = x + (size_t)b * Nc * Dc;
  for (int n = 0; n < Nc; ++n) s += xb[(size_t)n * Dc + tid];
  feat[tid] = s * (1.f / (float)Nc);
  for (int j = tid; j < TDc; j += 256) feat[Dc + j] = text[b * TDc + j];
  __syncthreads();
  float h = b1[tid];
  for (int k = 0; k < Dc + TDc; ++k) h += feat[k] * w1[k * Dc + tid];
  hid[tid] = fmaxf(h, 0.f);
  __syncthreads();
  if (tid < Ec) {
    float l = b2[tid];
    for (int j = 0; j < Dc; ++j) l += hid[j] * w2[j * Ec + tid];
    logits[tid] = l;
  }
  __syncthreads();
  if (tid == 0) {
    int i0 = 0;
    for (int e = 1; e < Ec; ++e) if (logits[e] > logits[i0]) i0 = e;
    int i1 = -1;
    for (int e = 0; e < Ec; ++e) {
      if (e == i0) continue;
      if (i1 < 0 || logits[e] > logits[i1]) i1 = e;
    }
    float m = fmaxf(logits[i0], logits[i1]);
    float e0 = __expf(logits[i0] - m), e1 = __expf(logits[i1] - m);
    float inv = 1.f / (e0 + e1);
    float p0 = e0 * inv, p1 = e1 * inv;
    for (int e = 0; e < Ec; ++e) cw[b * Ec + e] = 0.f;
    cw[b * Ec + i0] = p0; cw[b * Ec + i1] = p1;
    wout[b * 2 + 0] = p0; wout[b * 2 + 1] = p1;
  }
}

// ------------------------- generic WMMA GEMM -------------------------------
// C[M,N] = A[M,K] * B[K,N]; bf16 fragments, fp32 accumulation.
// EPI: 0 bf16 (acc+bias) | 1 bf16 gelu(acc+bias) | 3 fp32 (acc+bias)
//      4 bf16 (acc+bias+Res) | 2 d_out += cw[b,expert]*(acc+bias+Res)
template <int EPI, typename TA, typename TRes>
__global__ __launch_bounds__(128) void gemm_wmma_k(
    const TA* __restrict__ A, const float* __restrict__ Bw,
    void* __restrict__ Cout, const float* __restrict__ bias,
    const TRes* __restrict__ Res, float* __restrict__ Dacc,
    const float* __restrict__ cw, int expert,
    int Mdim, int Ndim, int Kdim) {
  const int tid = threadIdx.x;
  const int lane = tid & 31, wave = tid >> 5;
  const int wm = wave >> 1, wn = wave & 1;
  const int l15 = lane & 15, khalf = lane >> 4;
  const int m0 = blockIdx.y * 64, n0 = blockIdx.x * 64;

  v8f acc[2][2] = {};

#if HAVE_TDM
  // Double-buffered: TDM moves the A tile (async, TENSORcnt) while all waves
  // stage the transposed B tile and run WMMA on the previous buffers.
  __shared__ __align__(16) TA   Araw[2][64][32];
  __shared__ __align__(16) bf16 Bt[2][64][40];
  const int nk = Kdim >> 5;
  if (wave == 0)
    tdm_load_tile<TA>(&Araw[0][0][0], A + (size_t)m0 * Kdim, Kdim, Mdim - m0, Kdim);
  for (int i = tid; i < 32 * 64; i += 128) {
    int r = i >> 6, c = i & 63;
    Bt[0][c][r] = (bf16)Bw[(size_t)r * Ndim + (n0 + c)];
  }
  for (int it = 0; it < nk; ++it) {
    const int cur = it & 1, nxt = cur ^ 1;
    if (wave == 0) __builtin_amdgcn_s_wait_tensorcnt(0);
    __syncthreads();                       // publish Araw[cur] (TDM) + Bt[cur]
    if (it + 1 < nk) {
      const int k1 = (it + 1) * 32;
      if (wave == 0)
        tdm_load_tile<TA>(&Araw[nxt][0][0], A + (size_t)m0 * Kdim + k1,
                          Kdim, Mdim - m0, Kdim - k1);
      for (int i = tid; i < 32 * 64; i += 128) {
        int r = i >> 6, c = i & 63;
        Bt[nxt][c][r] = (bf16)Bw[(size_t)(k1 + r) * Ndim + (n0 + c)];
      }
    }
#pragma unroll
    for (int tm = 0; tm < 2; ++tm) {
      v16bf af;
      const int ar = wm * 32 + tm * 16 + l15;
#pragma unroll
      for (int j = 0; j < 16; ++j) {
        int kk = j + khalf * 16;
        af[j] = (bf16)(float)Araw[cur][ar][kk];
      }
#pragma unroll
      for (int tn = 0; tn < 2; ++tn) {
        const int bc = wn * 32 + tn * 16 + l15;
        v8bf blo = *(const v8bf*)&Bt[cur][bc][khalf * 16];
        v8bf bhi = *(const v8bf*)&Bt[cur][bc][khalf * 16 + 8];
        acc[tm][tn] = WMMA_BF16(af, mk16(blo, bhi), acc[tm][tn]);
      }
    }
  }
#else
  __shared__ __align__(16) bf16 As[64][40];
  __shared__ __align__(16) bf16 Bt[64][40];
  for (int k0 = 0; k0 < Kdim; k0 += 32) {
    for (int i = tid; i < 64 * 32; i += 128) {
      int r = i >> 5, c = i & 31;
      As[r][c] = (bf16)(float)A[(size_t)(m0 + r) * Kdim + (k0 + c)];
    }
    for (int i = tid; i < 32 * 64; i += 128) {
      int r = i >> 6, c = i & 63;
      Bt[c][r] = (bf16)Bw[(size_t)(k0 + r) * Ndim + (n0 + c)];
    }
    __syncthreads();
    if (k0 + 32 < Kdim) {
      __builtin_prefetch(&A[(size_t)(m0 + (tid >> 1)) * Kdim + (k0 + 32)], 0, 1);
      __builtin_prefetch(&Bw[(size_t)(k0 + 32 + (tid & 31)) * Ndim + n0], 0, 1);
    }
#pragma unroll
    for (int tm = 0; tm < 2; ++tm) {
      const int ar = wm * 32 + tm * 16 + l15;
      v8bf alo = *(const v8bf*)&As[ar][khalf * 8];
      v8bf ahi = *(const v8bf*)&As[ar][16 + khalf * 8];
      v16bf af = mk16(alo, ahi);
#pragma unroll
      for (int tn = 0; tn < 2; ++tn) {
        const int bc = wn * 32 + tn * 16 + l15;
        v8bf blo = *(const v8bf*)&Bt[bc][khalf * 8];
        v8bf bhi = *(const v8bf*)&Bt[bc][16 + khalf * 8];
        acc[tm][tn] = WMMA_BF16(af, mk16(blo, bhi), acc[tm][tn]);
      }
    }
    __syncthreads();
  }
#endif

#pragma unroll
  for (int tm = 0; tm < 2; ++tm)
#pragma unroll
    for (int tn = 0; tn < 2; ++tn) {
      const int colg = n0 + wn * 32 + tn * 16 + l15;
#pragma unroll
      for (int v = 0; v < 8; ++v) {
        const int rowg = m0 + wm * 32 + tm * 16 + v + 8 * khalf;
        float a = acc[tm][tn][v];
        if (bias) a += bias[colg];
        const size_t idx = (size_t)rowg * Ndim + colg;
        if constexpr (EPI == 0) {
          ((bf16*)Cout)[idx] = (bf16)a;
        } else if constexpr (EPI == 1) {
          ((bf16*)Cout)[idx] = (bf16)gelu_f(a);
        } else if constexpr (EPI == 3) {
          ((float*)Cout)[idx] = a;
        } else if constexpr (EPI == 4) {
          ((bf16*)Cout)[idx] = (bf16)(a + (float)Res[idx]);
        } else if constexpr (EPI == 2) {
          const float r = (float)Res[idx];
          const float sc = cw[(rowg >> 12) * Ec + expert];
          Dacc[idx] += sc * (a + r);
        }
      }
    }
}

// --------------------- FNO forward DFT (per b,d tile) ----------------------
__global__ __launch_bounds__(256) void fno_dft_fwd_k(
    const float* __restrict__ x, const float* __restrict__ tab,
    float* __restrict__ xftR, float* __restrict__ xftI) {
  const int bd = blockIdx.x;
  const int b = bd >> 8, d = bd & 255;
  const int tid = threadIdx.x;
  __shared__ float tile[64][65];
  __shared__ float SR[64][16], SI[64][16];
  const float* xb = x + (size_t)b * Nc * Dc + d;
  for (int i = tid; i < 4096; i += 256) tile[i >> 6][i & 63] = xb[(size_t)i * Dc];
  __syncthreads();
  const float* WcR = tab;          const float* WcI = tab + 1024;
  for (int i = tid; i < 1024; i += 256) {
    int h = i >> 4, ky = i & 15;
    float sr = 0.f, si = 0.f;
    for (int w = 0; w < 64; ++w) {
      float v = tile[h][w];
      sr += v * WcR[w * 16 + ky];
      si += v * WcI[w * 16 + ky];
    }
    SR[h][ky] = sr; SI[h][ky] = si;
  }
  __syncthreads();
  const float* WrR = tab + 2048;   const float* WrI = tab + 4096;
  for (int i = tid; i < 512; i += 256) {
    int slot = i >> 4, ky = i & 15;
    float tr = 0.f, ti = 0.f;
    for (int h = 0; h < 64; ++h) {
      float ar = WrR[slot * 64 + h], ai = WrI[slot * 64 + h];
      float sr = SR[h][ky], si = SI[h][ky];
      tr += ar * sr - ai * si;
      ti += ar * si + ai * sr;
    }
    size_t dst = ((size_t)(b * 256 + d) * 32 + slot) * 16 + ky;
    xftR[dst] = tr; xftI[dst] = ti;
  }
}

// -------------------- FNO spectral mix (WMMA complex GEMM) -----------------
__global__ __launch_bounds__(128) void fno_mode_mix_k(
    const float* __restrict__ xftR, const float* __restrict__ xftI,
    const float* __restrict__ w1r, const float* __restrict__ w1i,
    const float* __restrict__ w2r, const float* __restrict__ w2i,
    float* __restrict__ oftR, float* __restrict__ oftI) {
  const int mode = blockIdx.x;
  const int slot = mode >> 4, ky = mode & 15;
  const int kxw = (slot < 16) ? slot : (slot - 16);
  const float* WR = (slot < 16) ? w1r : w2r;
  const float* WI = (slot < 16) ? w1i : w2i;
  const int tid = threadIdx.x;

  __shared__ __align__(16) bf16 AsR[16][256];
  __shared__ __align__(16) bf16 AsI[16][256];
  for (int i = tid; i < 16 * 256; i += 128) {
    int bb = i >> 8, ch = i & 255;
    float vr = 0.f, vi = 0.f;
    if (bb < Bc) {
      size_t src = ((size_t)(bb * 256 + ch) * 32 + slot) * 16 + ky;
      vr = xftR[src]; vi = xftI[src];
    }
    AsR[bb][ch] = (bf16)vr; AsI[bb][ch] = (bf16)vi;
  }
  __syncthreads();

  const int lane = tid & 31, wave = tid >> 5;
  const int l15 = lane & 15, khalf = lane >> 4;
  const int nbase = wave * 64;

  v8f accR[4] = {}; v8f accI[4] = {};
  for (int k0 = 0; k0 < 256; k0 += 32) {
    v8bf rlo = *(const v8bf*)&AsR[l15][k0 + khalf * 8];
    v8bf rhi = *(const v8bf*)&AsR[l15][k0 + 16 + khalf * 8];
    v8bf ilo = *(const v8bf*)&AsI[l15][k0 + khalf * 8];
    v8bf ihi = *(const v8bf*)&AsI[l15][k0 + 16 + khalf * 8];
    v16bf ar = mk16(rlo, rhi);
    v16bf ai = mk16(ilo, ihi);
    v16bf ain;
#pragma unroll
    for (int j = 0; j < 16; ++j) ain[j] = (bf16)(-(float)ai[j]);
#pragma unroll
    for (int tn = 0; tn < 4; ++tn) {
      const int o = nbase + tn * 16 + l15;
      v16bf br, bi;
#pragma unroll
      for (int j = 0; j < 16; ++j) {
        int ich = k0 + (j < 8 ? j : j + 8) + khalf * 8;
        size_t adr = ((size_t)ich * 256 + o) * 256 + kxw * 16 + ky;
        br[j] = (bf16)WR[adr];
        bi[j] = (bf16)WI[adr];
      }
      accR[tn] = WMMA_BF16(ar, br, accR[tn]);
      accR[tn] = WMMA_BF16(ain, bi, accR[tn]);
      accI[tn] = WMMA_BF16(ar, bi, accI[tn]);
      accI[tn] = WMMA_BF16(ai, br, accI[tn]);
    }
  }
  if (khalf == 0) {
    for (int tn = 0; tn < 4; ++tn) {
      const int o = nbase + tn * 16 + l15;
#pragma unroll
      for (int v = 0; v < 8; ++v) {   // v == batch index (rows 8..15 were zero)
        size_t dst = ((size_t)(v * 256 + o) * 32 + slot) * 16 + ky;
        oftR[dst] = accR[tn][v];
        oftI[dst] = accI[tn][v];
      }
    }
  }
}

// ---------------- FNO inverse DFT + gelu + residual + MoE scale ------------
__global__ __launch_bounds__(256) void fno_idft_combine_k(
    const float* __restrict__ oftR, const float* __restrict__ oftI,
    const float* __restrict__ tab, const float* __restrict__ x2,
    const float* __restrict__ x, const float* __restrict__ cw,
    float* __restrict__ out) {
  const int bo = blockIdx.x;
  const int b = bo >> 8, o = bo & 255;
  const int tid = threadIdx.x;
  __shared__ float OR_[32][16], OI_[32][16];
  __shared__ float GR[64][16], GI[64][16];
  const size_t srcb = (size_t)(b * 256 + o) * 512;
  for (int i = tid; i < 512; i += 256) {
    OR_[i >> 4][i & 15] = oftR[srcb + i];
    OI_[i >> 4][i & 15] = oftI[srcb + i];
  }
  __syncthreads();
  const float* UrR = tab + 6144; const float* UrI = tab + 8192;
  for (int i = tid; i < 1024; i += 256) {
    int h = i >> 4, ky = i & 15;
    float gr = 0.f, gi = 0.f;
    for (int s = 0; s < 32; ++s) {
      float ur = UrR[h * 32 + s], ui = UrI[h * 32 + s];
      float orv = OR_[s][ky], oiv = OI_[s][ky];
      gr += ur * orv - ui * oiv;
      gi += ur * oiv + ui * orv;
    }
    GR[h][ky] = gr; GI[h][ky] = gi;
  }
  __syncthreads();
  const float* UcR = tab + 10240; const float* UcI = tab + 11264;
  const float sc = cw[b * Ec + 0];
  for (int i = tid; i < 4096; i += 256) {
    int h = i >> 6, w = i & 63;
    float s = 0.f;
    for (int ky = 0; ky < 16; ++ky)
      s += GR[h][ky] * UcR[ky * 64 + w] - GI[h][ky] * UcI[ky * 64 + w];
    s *= (1.f / 4096.f);
    const int n = h * 64 + w;
    const size_t idx = ((size_t)(b * Nc + n)) * Dc + o;
    float val = gelu_f(s + x2[idx]) + x[idx];
    out[idx] = sc * val;   // first expert initializes d_out
  }
}

// ------------------------------ layernorm ----------------------------------
template <typename T>
__global__ __launch_bounds__(256) void layernorm_k(
    const T* __restrict__ X, const float* __restrict__ g,
    const float* __restrict__ bb, bf16* __restrict__ Y) {
  const int row = blockIdx.x, tid = threadIdx.x;
  __shared__ float red[256];
  float v = (float)X[(size_t)row * Dc + tid];
  red[tid] = v; __syncthreads();
  for (int s = 128; s > 0; s >>= 1) { if (tid < s) red[tid] += red[tid + s]; __syncthreads(); }
  const float mu = red[0] * (1.f / (float)Dc);
  __syncthreads();
  const float d = v - mu;
  red[tid] = d * d; __syncthreads();
  for (int s = 128; s > 0; s >>= 1) { if (tid < s) red[tid] += red[tid + s]; __syncthreads(); }
  const float var = red[0] * (1.f / (float)Dc);
  Y[(size_t)row * Dc + tid] = (bf16)(d * rsqrtf(var + 1e-5f) * g[tid] + bb[tid]);
}

// ------------------- window attention (per window x head) ------------------
__global__ __launch_bounds__(128) void window_attn_k(
    const bf16* __restrict__ qkv, bf16* __restrict__ ao) {
  const int blk = blockIdx.x;          // 4096 = 512 windows * 8 heads
  const int head = blk & 7;
  const int win = blk >> 3;
  const int b = win >> 6;
  const int widx = win & 63;
  const int wy = widx >> 3, wx = widx & 7;
  __shared__ __align__(16) bf16 Qs[64][40], Ks[64][40];
  __shared__ __align__(16) bf16 Vt[32][72];       // V transposed: [dh][token]
  __shared__ float Sf[64][65];
  __shared__ __align__(16) bf16 Ps[64][72];
  const int tid = threadIdx.x;
  for (int i = tid; i < 64 * 32; i += 128) {
    int l = i >> 5, c = i & 31;
    int iy = l >> 3, ix = l & 7;
    int n = (wy * 8 + iy) * 64 + (wx * 8 + ix);
    size_t row = (size_t)(b * Nc + n) * (3 * Dc);
    int col = head * DHc + c;
    Qs[l][c] = qkv[row + col];
    Ks[l][c] = qkv[row + Dc + col];
    Vt[c][l] = qkv[row + 2 * Dc + col];
  }
  __syncthreads();
  const int lane = tid & 31, wave = tid >> 5;
  const int l15 = lane & 15, khalf = lane >> 4;

  // S = Q K^T / sqrt(dh)
  {
    const int qr = wave * 16 + l15;
    v8bf qlo = *(const v8bf*)&Qs[qr][khalf * 8];
    v8bf qhi = *(const v8bf*)&Qs[qr][16 + khalf * 8];
    v16bf qf = mk16(qlo, qhi);
#pragma unroll
    for (int tn = 0; tn < 4; ++tn) {
      const int kn = tn * 16 + l15;
      v8bf klo = *(const v8bf*)&Ks[kn][khalf * 8];
      v8bf khi = *(const v8bf*)&Ks[kn][16 + khalf * 8];
      v8f s = {};
      s = WMMA_BF16(qf, mk16(klo, khi), s);
#pragma unroll
      for (int v = 0; v < 8; ++v) {
        int r = wave * 16 + v + 8 * khalf;
        Sf[r][tn * 16 + l15] = s[v] * 0.17677669529663688f;
      }
    }
  }
  __syncthreads();
  // softmax rows
  if (tid < 64) {
    float mx = -1e30f;
    for (int j = 0; j < 64; ++j) mx = fmaxf(mx, Sf[tid][j]);
    float sum = 0.f;
    for (int j = 0; j < 64; ++j) sum += __expf(Sf[tid][j] - mx);
    const float inv = 1.f / sum;
    for (int j = 0; j < 64; ++j) Ps[tid][j] = (bf16)(__expf(Sf[tid][j] - mx) * inv);
  }
  __syncthreads();
  // O = P @ V   (M=64, K=64, N=32)
  v8f oacc[2] = {};
  for (int ks = 0; ks < 2; ++ks) {
    const int pr = wave * 16 + l15;
    v8bf plo = *(const v8bf*)&Ps[pr][ks * 32 + khalf * 8];
    v8bf phi = *(const v8bf*)&Ps[pr][ks * 32 + 16 + khalf * 8];
    v16bf pf = mk16(plo, phi);
#pragma unroll
    for (int tn = 0; tn < 2; ++tn) {
      const int vc = tn * 16 + l15;
      v8bf vlo = *(const v8bf*)&Vt[vc][ks * 32 + khalf * 8];
      v8bf vhi = *(const v8bf*)&Vt[vc][ks * 32 + 16 + khalf * 8];
      oacc[tn] = WMMA_BF16(pf, mk16(vlo, vhi), oacc[tn]);
    }
  }
#pragma unroll
  for (int tn = 0; tn < 2; ++tn)
#pragma unroll
    for (int v = 0; v < 8; ++v) {
      int l = wave * 16 + v + 8 * khalf;
      int iy = l >> 3, ix = l & 7;
      int n = (wy * 8 + iy) * 64 + (wx * 8 + ix);
      ao[(size_t)(b * Nc + n) * Dc + head * DHc + tn * 16 + l15] = (bf16)oacc[tn][v];
    }
}

// ------------------------------- launcher ----------------------------------
extern "C" void kernel_launch(void* const* d_in, const int* in_sizes, int n_in,
                              void* d_out, int out_size, void* d_ws, size_t ws_size,
                              hipStream_t stream) {
  const float* x        = (const float*)d_in[0];
  const float* text     = (const float*)d_in[1];
  const float* gate_w1  = (const float*)d_in[2];
  const float* gate_b1  = (const float*)d_in[3];
  const float* gate_w2  = (const float*)d_in[4];
  const float* gate_b2  = (const float*)d_in[5];
  const float* fno_w1r  = (const float*)d_in[6];
  const float* fno_w1i  = (const float*)d_in[7];
  const float* fno_w2r  = (const float*)d_in[8];
  const float* fno_w2i  = (const float*)d_in[9];
  const float* conv_w   = (const float*)d_in[10];
  const float* conv_b   = (const float*)d_in[11];
  const float* mlp1_w1  = (const float*)d_in[12];
  const float* mlp1_b1  = (const float*)d_in[13];
  const float* mlp1_w2  = (const float*)d_in[14];
  const float* mlp1_b2  = (const float*)d_in[15];
  const float* mlp2_w1  = (const float*)d_in[16];
  const float* mlp2_b1  = (const float*)d_in[17];
  const float* mlp2_w2  = (const float*)d_in[18];
  const float* mlp2_b2  = (const float*)d_in[19];
  const float* ln1_g    = (const float*)d_in[20];
  const float* ln1_b    = (const float*)d_in[21];
  const float* qkv_w    = (const float*)d_in[22];
  const float* qkv_b    = (const float*)d_in[23];
  const float* out_w    = (const float*)d_in[24];
  const float* out_b    = (const float*)d_in[25];
  const float* ln2_g    = (const float*)d_in[26];
  const float* ln2_b    = (const float*)d_in[27];
  const float* wa_w1    = (const float*)d_in[28];
  const float* wa_b1    = (const float*)d_in[29];
  const float* wa_w2    = (const float*)d_in[30];
  const float* wa_b2    = (const float*)d_in[31];

  float* fws  = (float*)d_ws;
  float* tab  = fws + OFF_TAB;
  float* cw   = fws + OFF_CW;
  float* xftR = fws + F_XFTR;
  float* xftI = fws + F_XFTI;
  float* oftR = fws + F_OFTR;
  float* oftI = fws + F_OFTI;
  float* x2   = fws + F_X2;

  const size_t bbase = ((F_END * sizeof(float)) + 255) & ~(size_t)255;
  bf16* bws  = (bf16*)((char*)d_ws + bbase);
  bf16* hid  = bws + H_HID;
  bf16* qkvs = bws + H_QKV;
  bf16* lnh  = bws + H_LN;
  bf16* ao   = bws + H_AO;
  bf16* xa   = bws + H_XA;

  float* out   = (float*)d_out;
  float* wghts = out + (size_t)Bc * Nc * Dc;

  const int MT = Bc * Nc;  // 32768 token rows

  fill_tables_k<<<1, 256, 0, stream>>>(tab);
  gating_k<<<Bc, 256, 0, stream>>>(x, text, gate_w1, gate_b1, gate_w2, gate_b2, cw, wghts);

  // ---- expert 0: FNO (initializes d_out with cw[b,0]*e0) ----
  gemm_wmma_k<3, float, float><<<dim3(4, 512), 128, 0, stream>>>(
      x, conv_w, x2, conv_b, (const float*)nullptr, nullptr, nullptr, 0, MT, Dc, Dc);
  fno_dft_fwd_k<<<Bc * Dc, 256, 0, stream>>>(x, tab, xftR, xftI);
  fno_mode_mix_k<<<512, 128, 0, stream>>>(xftR, xftI, fno_w1r, fno_w1i, fno_w2r, fno_w2i, oftR, oftI);
  fno_idft_combine_k<<<Bc * Dc, 256, 0, stream>>>(oftR, oftI, tab, x2, x, cw, out);

  // ---- expert 1: MLP ----
  gemm_wmma_k<1, float, float><<<dim3(16, 512), 128, 0, stream>>>(
      x, mlp1_w1, hid, mlp1_b1, (const float*)nullptr, nullptr, nullptr, 0, MT, HIDc, Dc);
  gemm_wmma_k<2, bf16, float><<<dim3(4, 512), 128, 0, stream>>>(
      hid, mlp1_w2, nullptr, mlp1_b2, x, out, cw, 1, MT, Dc, HIDc);

  // ---- expert 2: window attention ----
  layernorm_k<float><<<MT, 256, 0, stream>>>(x, ln1_g, ln1_b, lnh);
  gemm_wmma_k<0, bf16, float><<<dim3(12, 512), 128, 0, stream>>>(
      lnh, qkv_w, qkvs, qkv_b, (const float*)nullptr, nullptr, nullptr, 0, MT, 3 * Dc, Dc);
  window_attn_k<<<Bc * 64 * NHc, 128, 0, stream>>>(qkvs, ao);
  gemm_wmma_k<4, bf16, float><<<dim3(4, 512), 128, 0, stream>>>(
      ao, out_w, xa, out_b, x, nullptr, nullptr, 0, MT, Dc, Dc);
  layernorm_k<bf16><<<MT, 256, 0, stream>>>(xa, ln2_g, ln2_b, lnh);
  gemm_wmma_k<1, bf16, float><<<dim3(16, 512), 128, 0, stream>>>(
      lnh, wa_w1, hid, wa_b1, (const float*)nullptr, nullptr, nullptr, 0, MT, HIDc, Dc);
  gemm_wmma_k<2, bf16, bf16><<<dim3(4, 512), 128, 0, stream>>>(
      hid, wa_w2, nullptr, wa_b2, xa, out, cw, 2, MT, Dc, HIDc);

  // ---- expert 3: MLP ----
  gemm_wmma_k<1, float, float><<<dim3(16, 512), 128, 0, stream>>>(
      x, mlp2_w1, hid, mlp2_b1, (const float*)nullptr, nullptr, nullptr, 0, MT, HIDc, Dc);
  gemm_wmma_k<2, bf16, float><<<dim3(4, 512), 128, 0, stream>>>(
      hid, mlp2_w2, nullptr, mlp2_b2, x, out, cw, 3, MT, Dc, HIDc);
}
